// MessagePassingCell_3779571220745
// MI455X (gfx1250) — matrix operand order repaired and checked by reading, not compiled, not run on hardware
//
#include <hip/hip_runtime.h>

// ---------- problem constants (from the reference) ----------
#define Gg   4096
#define Nn   524288
#define Hh   128
#define NT   8            // 128 / 16 output col tiles
#define TM   128          // rows per block (4 waves x 32 rows)
#define LDA  136          // padded LDS stride (128 + 8 bf16): 272B/row = 68 dwords = 4 mod 64 banks
#define KT1  8            // layer-1 K tiles (K=256)
#define KT2  4            // layer-2 K tiles (K=128)

typedef __attribute__((ext_vector_type(16))) __bf16 v16bf;
typedef __attribute__((ext_vector_type(8)))  __bf16 v8bf;
typedef __attribute__((ext_vector_type(4)))  __bf16 v4bf;
typedef __attribute__((ext_vector_type(8)))  float  v8f;

__device__ __forceinline__ __bf16 f2bf(float f) {
    return (__bf16)f;   // native f32->bf16 cvt (RNE) on gfx1250
}

__device__ __forceinline__ v8f wmma_bf16(v16bf a, v16bf b, v8f c) {
    return __builtin_amdgcn_wmma_f32_16x16x32_bf16(
        false, a, false, b, (short)0, c, false, false);
}

// 16x32 bf16 A fragment (ISA 16-bit A layout) from row-major LDS tile.
__device__ __forceinline__ v16bf load_afrag(const __bf16* base, int lane, int kOff) {
    const __bf16* p = base + (lane & 15) * LDA + kOff + ((lane >> 4) * 8);
    v8bf lo = *(const v8bf*)p;
    v8bf hi = *(const v8bf*)(p + 16);
    v16bf r;
#pragma unroll
    for (int i = 0; i < 8; ++i) { r[i] = lo[i]; r[8 + i] = hi[i]; }
    return r;
}

// ---------- kernel 1: pack weights [K,128] f32 -> bf16 B-fragments ----------
// frag f = kt*NT + nt ; k = kt*32 + (e<8 ? 8*(lane>>4)+e : 16+8*(lane>>4)+e-8) ; n = nt*16 + (lane&15)
__global__ void pack_w(const float* __restrict__ W, __bf16* __restrict__ dst, int KT) {
    int total = KT * NT * 512;
    for (int i = blockIdx.x * blockDim.x + threadIdx.x; i < total; i += gridDim.x * blockDim.x) {
        int f    = i >> 9;
        int lane = (i >> 4) & 31;
        int e    = i & 15;
        int kt = f >> 3, nt = f & 7;
        int k = kt * 32 + ((lane >> 4) * 8) + (e & 7) + ((e >> 3) * 16);
        int n = nt * 16 + (lane & 15);
        dst[i] = f2bf(W[k * Hh + n]);
    }
}

// ---------- kernel 2: segment boundaries (ids sorted -> binary search) ----------
__global__ void seg_start_kernel(const int* __restrict__ seg, int* __restrict__ start) {
    int g = blockIdx.x * blockDim.x + threadIdx.x;
    if (g > Gg) return;
    if (g == Gg) { start[Gg] = Nn; return; }
    int lo = 0, hi = Nn;
    while (lo < hi) {
        int mid = (lo + hi) >> 1;
        if (seg[mid] < g) lo = mid + 1; else hi = mid;
    }
    start[g] = lo;
}

// ---------- kernel 3: segment sum, one block per graph ----------
__global__ __launch_bounds__(Hh) void agg_kernel(const float* __restrict__ node,
                                                 const int* __restrict__ start,
                                                 float* __restrict__ agg) {
    int g = blockIdx.x;
    int c = threadIdx.x;
    int s0 = start[g], s1 = start[g + 1];
    float acc = 0.f;
    for (int r = s0; r < s1; ++r) acc += node[r * Hh + c];
    agg[g * Hh + c] = acc;
}

// ---------- kernel 4/5: fused 2-layer MLP (2H -> H -> H, ReLU) ----------
// 128 rows/block, 4 waves x 32 rows. Layer-1 K split into X0-half then X1-half,
// single LDS tile reused for X0 / X1 / h1. B fragments double-buffered in LDS.
__global__ __launch_bounds__(128) void mlp_kernel(const float* __restrict__ X0,
                                                  const float* __restrict__ X1,
                                                  const int* __restrict__ gather,
                                                  const __bf16* __restrict__ W1,
                                                  const float* __restrict__ b1,
                                                  const __bf16* __restrict__ W2,
                                                  const float* __restrict__ b2,
                                                  float* __restrict__ out) {
    __shared__ __bf16 Atile[TM * LDA];       // 34816 B (X0 tile -> X1 tile -> h1 tile)
    __shared__ __bf16 Bbuf[2][NT * 512];     // 2 x 8192 B, double-buffered B fragments
    __shared__ int    segl[TM];

    const int tid  = threadIdx.x;
    const int lane = tid & 31;
    const int wave = tid >> 5;
    const int rowbase = blockIdx.x * TM;
    const int ncol  = lane & 15;
    const int mhalf = (lane >> 4) * 8;

    // cooperative B staging: one k-step (8 frags = 8KB) as 512 uint4
    auto stageB = [&](const __bf16* W, int kt, __bf16* dst) {
        const uint4* s = (const uint4*)(W + (size_t)kt * NT * 512);
        uint4* d = (uint4*)dst;
#pragma unroll
        for (int j = 0; j < 4; ++j) d[tid + 128 * j] = s[tid + 128 * j];
    };

    // vectorized A staging: float4 in, packed 4x bf16 (b64) out; 32 iters/thread
    auto stageA = [&](const float* src, bool useGather) {
#pragma unroll 4
        for (int i = tid; i < TM * 32; i += 128) {
            int r  = i >> 5;
            int c4 = (i & 31) << 2;
            const float* row = useGather ? (src + (size_t)segl[r] * Hh)
                                         : (src + (size_t)(rowbase + r) * Hh);
            float4 v = *(const float4*)(row + c4);
            v4bf o;
            o[0] = f2bf(v.x); o[1] = f2bf(v.y); o[2] = f2bf(v.z); o[3] = f2bf(v.w);
            *(v4bf*)(&Atile[r * LDA + c4]) = o;
        }
    };

    segl[tid] = gather ? gather[rowbase + tid] : (rowbase + tid);

    stageA(X0, false);
    stageB(W1, 0, Bbuf[0]);
    __syncthreads();

    const __bf16* Ar0 = &Atile[(wave * 32) * LDA];
    const __bf16* Ar1 = &Atile[(wave * 32 + 16) * LDA];

    v8f acc[2][NT];
#pragma unroll
    for (int nt = 0; nt < NT; ++nt) {
        float bv = b1[nt * 16 + ncol];
#pragma unroll
        for (int v = 0; v < 8; ++v) { acc[0][nt][v] = bv; acc[1][nt][v] = bv; }
    }

    // ---- layer 1, pass 0: K = 0..127 (X0 half), W1 k-tiles 0..3 ----
#pragma unroll
    for (int kt = 0; kt < 4; ++kt) {
        if (kt < 3) stageB(W1, kt + 1, Bbuf[(kt + 1) & 1]);
        const __bf16* Bl = Bbuf[kt & 1];
        v16bf a0 = load_afrag(Ar0, lane, kt * 32);
        v16bf a1 = load_afrag(Ar1, lane, kt * 32);
#pragma unroll
        for (int nt = 0; nt < NT; ++nt) {
            v16bf b = *(const v16bf*)(Bl + nt * 512 + lane * 16);
            acc[0][nt] = wmma_bf16(a0, b, acc[0][nt]);
            acc[1][nt] = wmma_bf16(a1, b, acc[1][nt]);
        }
        __syncthreads();
    }

    // boundary: restage tile with X1 half (gathered), stage W1 k-tile 4
    stageA(X1, gather != nullptr || true /* X1 indexed via segl (identity if no gather) */);
    stageB(W1, 4, Bbuf[0]);
    __syncthreads();

    // ---- layer 1, pass 1: K = 128..255 (X1 half), W1 k-tiles 4..7 ----
#pragma unroll
    for (int kt = 0; kt < 4; ++kt) {
        if (kt < 3) stageB(W1, 5 + kt, Bbuf[(kt + 1) & 1]);
        const __bf16* Bl = Bbuf[kt & 1];
        v16bf a0 = load_afrag(Ar0, lane, kt * 32);
        v16bf a1 = load_afrag(Ar1, lane, kt * 32);
#pragma unroll
        for (int nt = 0; nt < NT; ++nt) {
            v16bf b = *(const v16bf*)(Bl + nt * 512 + lane * 16);
            acc[0][nt] = wmma_bf16(a0, b, acc[0][nt]);
            acc[1][nt] = wmma_bf16(a1, b, acc[1][nt]);
        }
        __syncthreads();
    }

    // ReLU -> bf16 -> h1 into the same tile (wave-local rows, no barrier needed)
    __bf16* Hrow = &Atile[(wave * 32) * LDA];
#pragma unroll
    for (int sub = 0; sub < 2; ++sub)
#pragma unroll
        for (int nt = 0; nt < NT; ++nt)
#pragma unroll
            for (int v = 0; v < 8; ++v) {
                float x = acc[sub][nt][v];
                x = x > 0.f ? x : 0.f;
                Hrow[(sub * 16 + mhalf + v) * LDA + nt * 16 + ncol] = f2bf(x);
            }
    stageB(W2, 0, Bbuf[0]);
    __syncthreads();

    // ---- layer 2: K = 0..127 (h1), W2 k-tiles 0..3 ----
    v8f acc2[2][NT];
#pragma unroll
    for (int nt = 0; nt < NT; ++nt) {
        float bv = b2[nt * 16 + ncol];
#pragma unroll
        for (int v = 0; v < 8; ++v) { acc2[0][nt][v] = bv; acc2[1][nt][v] = bv; }
    }
#pragma unroll
    for (int kt = 0; kt < KT2; ++kt) {
        if (kt < 3) stageB(W2, kt + 1, Bbuf[(kt + 1) & 1]);
        const __bf16* Bl = Bbuf[kt & 1];
        v16bf a0 = load_afrag(Ar0, lane, kt * 32);
        v16bf a1 = load_afrag(Ar1, lane, kt * 32);
#pragma unroll
        for (int nt = 0; nt < NT; ++nt) {
            v16bf b = *(const v16bf*)(Bl + nt * 512 + lane * 16);
            acc2[0][nt] = wmma_bf16(a0, b, acc2[0][nt]);
            acc2[1][nt] = wmma_bf16(a1, b, acc2[1][nt]);
        }
        __syncthreads();
    }

    // ReLU -> global
#pragma unroll
    for (int sub = 0; sub < 2; ++sub)
#pragma unroll
        for (int nt = 0; nt < NT; ++nt)
#pragma unroll
            for (int v = 0; v < 8; ++v) {
                float x = acc2[sub][nt][v];
                x = x > 0.f ? x : 0.f;
                out[(size_t)(rowbase + wave * 32 + sub * 16 + mhalf + v) * Hh
                    + nt * 16 + ncol] = x;
            }
}

extern "C" void kernel_launch(void* const* d_in, const int* in_sizes, int n_in,
                              void* d_out, int out_size, void* d_ws, size_t ws_size,
                              hipStream_t stream) {
    const float* updated_set  = (const float*)d_in[0];
    const float* updated_node = (const float*)d_in[1];
    const int*   segment_ids  = (const int*)  d_in[2];
    const float* Ws1 = (const float*)d_in[3];
    const float* bs1 = (const float*)d_in[4];
    const float* Ws2 = (const float*)d_in[5];
    const float* bs2 = (const float*)d_in[6];
    const float* Wn1 = (const float*)d_in[7];
    const float* bn1 = (const float*)d_in[8];
    const float* Wn2 = (const float*)d_in[9];
    const float* bn2 = (const float*)d_in[10];

    float* s_out = (float*)d_out;                   // [G,128]
    float* n_out = (float*)d_out + (size_t)Gg * Hh; // [N,128]

    char* ws = (char*)d_ws;
    size_t off = 0;
    float* nodes_agg = (float*)(ws + off); off += (size_t)Gg * Hh * 4;
    int*   seg_start = (int*)(ws + off);   off += ((size_t)(Gg + 1) * 4 + 63) & ~63ull;
    __bf16* pWs1 = (__bf16*)(ws + off); off += (size_t)KT1 * NT * 512 * 2;
    __bf16* pWs2 = (__bf16*)(ws + off); off += (size_t)KT2 * NT * 512 * 2;
    __bf16* pWn1 = (__bf16*)(ws + off); off += (size_t)KT1 * NT * 512 * 2;
    __bf16* pWn2 = (__bf16*)(ws + off); off += (size_t)KT2 * NT * 512 * 2;
    (void)ws_size; (void)in_sizes; (void)n_in; (void)out_size;

    pack_w<<<128, 256, 0, stream>>>(Ws1, pWs1, KT1);
    pack_w<<<64,  256, 0, stream>>>(Ws2, pWs2, KT2);
    pack_w<<<128, 256, 0, stream>>>(Wn1, pWn1, KT1);
    pack_w<<<64,  256, 0, stream>>>(Wn2, pWn2, KT2);

    seg_start_kernel<<<(Gg + 1 + 255) / 256, 256, 0, stream>>>(segment_ids, seg_start);
    agg_kernel<<<Gg, Hh, 0, stream>>>(updated_node, seg_start, nodes_agg);

    mlp_kernel<<<Gg / TM, 128, 0, stream>>>(nodes_agg, updated_set, nullptr,
                                            pWs1, bs1, pWs2, bs2, s_out);
    mlp_kernel<<<Nn / TM, 128, 0, stream>>>(updated_node, s_out, segment_ids,
                                            pWn1, bn1, pWn2, bn2, n_out);
}
